// TopologicalQualiaLoss_15513421873467
// MI455X (gfx1250) — matrix operand order repaired and checked by reading, not compiled, not run on hardware
//
#include <hip/hip_runtime.h>
#include <hip/hip_bf16.h>
#include <math.h>

#define TPTS 8192
#define DIM  512
#define KNN  5
#define CSPLIT 4
#define WPB 2
#define COLS_PER_SPLIT (TPTS / CSPLIT)

typedef __attribute__((ext_vector_type(16))) __bf16 v16bf;
typedef __attribute__((ext_vector_type(8)))  float  v8f;

// ---------------- prep: sq[i] = sum(x^2), bf16 hi/lo split of X ----------------
__global__ __launch_bounds__(256) void prep_kernel(const float* __restrict__ x,
                                                   unsigned short* __restrict__ xhi,
                                                   unsigned short* __restrict__ xlo,
                                                   float* __restrict__ sq) {
  const int wave = threadIdx.x >> 5;
  const int lane = threadIdx.x & 31;
  const int row  = blockIdx.x * 8 + wave;
  const float* xr = x + (size_t)row * DIM;
  float ss = 0.f;
#pragma unroll
  for (int c = 0; c < 4; ++c) {
    const int base = (c * 32 + lane) * 4;
    const float4 v = *(const float4*)(xr + base);
    const float f[4] = {v.x, v.y, v.z, v.w};
    unsigned short hu[4], lu[4];
#pragma unroll
    for (int e = 0; e < 4; ++e) {
      ss = fmaf(f[e], f[e], ss);
      __hip_bfloat16 hb = __float2bfloat16(f[e]);
      float resid = f[e] - __bfloat162float(hb);
      __hip_bfloat16 lb = __float2bfloat16(resid);
      __builtin_memcpy(&hu[e], &hb, 2);
      __builtin_memcpy(&lu[e], &lb, 2);
    }
    *(ushort4*)(xhi + (size_t)row * DIM + base) = make_ushort4(hu[0], hu[1], hu[2], hu[3]);
    *(ushort4*)(xlo + (size_t)row * DIM + base) = make_ushort4(lu[0], lu[1], lu[2], lu[3]);
  }
#pragma unroll
  for (int off = 16; off > 0; off >>= 1) ss += __shfl_down(ss, off, 32);
  if (lane == 0) sq[row] = ss;
}

// ---------------- main: fused bf16x3 WMMA Gram + streaming top-5 ----------------
__global__ __launch_bounds__(64) void knn_kernel(const unsigned short* __restrict__ XH,
                                                 const unsigned short* __restrict__ XL,
                                                 const float* __restrict__ sq,
                                                 float* __restrict__ knnbuf) {
  __shared__ float ltile[WPB][16 * 17];
  const int tid  = threadIdx.x;
  const int wave = tid >> 5;
  const int lane = tid & 31;
  const int h    = lane >> 4;        // lane half
  const int m    = lane & 15;        // row (A) / col (B,C)
  const int rblk = blockIdx.x / CSPLIT;
  const int cs   = blockIdx.x % CSPLIT;
  const int R0   = rblk * (WPB * 16) + wave * 16;
  const int Cbeg = cs * COLS_PER_SPLIT;
  const int Cend = Cbeg + COLS_PER_SPLIT;

  // A fragment addressing (16-bit A 16x32): lane half h holds K chunks {h*8.., 16+h*8..}
  const unsigned short* arh = XH + (size_t)(R0 + m) * DIM + h * 8;
  const unsigned short* arl = XL + (size_t)(R0 + m) * DIM + h * 8;

  float sqa[8];
#pragma unroll
  for (int v = 0; v < 8; ++v) sqa[v] = sq[R0 + 8 * h + v];   // C layout: M = v + 8h

  float t0 = 3.0e38f, t1 = 3.0e38f, t2 = 3.0e38f, t3 = 3.0e38f, t4 = 3.0e38f;
  float* tp = ltile[wave];

  for (int C0 = Cbeg; C0 < Cend; C0 += 16) {
    // B fragment addressing (16-bit B 32x16): lane half h holds K chunks {h*8.., 16+h*8..}
    const unsigned short* brh = XH + (size_t)(C0 + m) * DIM + h * 8;
    const unsigned short* brl = XL + (size_t)(C0 + m) * DIM + h * 8;
    const float sqb = sq[C0 + m];

    v8f accHH = {};  // hi*hi chain
    v8f accX  = {};  // hi*lo + lo*hi chain
#pragma unroll 4
    for (int k0 = 0; k0 < DIM; k0 += 32) {
      v16bf ah, al, bh, bl;
      uint4* pa;
      pa = (uint4*)&ah; pa[0] = *(const uint4*)(arh + k0); pa[1] = *(const uint4*)(arh + k0 + 16);
      pa = (uint4*)&al; pa[0] = *(const uint4*)(arl + k0); pa[1] = *(const uint4*)(arl + k0 + 16);
      pa = (uint4*)&bh; pa[0] = *(const uint4*)(brh + k0); pa[1] = *(const uint4*)(brh + k0 + 16);
      pa = (uint4*)&bl; pa[0] = *(const uint4*)(brl + k0); pa[1] = *(const uint4*)(brl + k0 + 16);
      accHH = __builtin_amdgcn_wmma_f32_16x16x32_bf16(false, ah, false, bh, (short)0, accHH, false, false);
      accX  = __builtin_amdgcn_wmma_f32_16x16x32_bf16(false, ah, false, bl, (short)0, accX,  false, false);
      accX  = __builtin_amdgcn_wmma_f32_16x16x32_bf16(false, al, false, bh, (short)0, accX,  false, false);
    }

    // distances^2 for this 16x16 tile -> LDS
#pragma unroll
    for (int v = 0; v < 8; ++v) {
      const int Mg = R0 + v + 8 * h;
      float d2 = sqa[v] + sqb - 2.0f * (accHH[v] + accX[v]);
      d2 = fmaxf(d2, 0.0f);
      if (Mg == C0 + m) d2 = 0.0f;                 // exact zero on diagonal
      tp[(v + 8 * h) * 17 + m] = d2;
    }
    __syncthreads();

    // lane (row m, half h) scans its 8 columns; branchless sorted-5 insert
    const float* rp = tp + m * 17 + h * 8;
#pragma unroll
    for (int c = 0; c < 8; ++c) {
      const float xx = rp[c];
      const float n0 = fminf(t0, xx); const float c1 = fmaxf(t0, xx);
      const float n1 = fminf(t1, c1); const float c2 = fmaxf(t1, c1);
      const float n2 = fminf(t2, c2); const float c3 = fmaxf(t2, c2);
      const float n3 = fminf(t3, c3); const float c4 = fmaxf(t3, c3);
      const float n4 = fminf(t4, c4);
      t0 = n0; t1 = n1; t2 = n2; t3 = n3; t4 = n4;
    }
    __syncthreads();
  }

  // merge the two lane-halves of each row through LDS, emit top-5 d^2 per row
  tp[lane * 5 + 0] = t0; tp[lane * 5 + 1] = t1; tp[lane * 5 + 2] = t2;
  tp[lane * 5 + 3] = t3; tp[lane * 5 + 4] = t4;
  __syncthreads();
  if (lane < 16) {
    float cand[10];
#pragma unroll
    for (int i = 0; i < 5; ++i) { cand[i] = tp[m * 5 + i]; cand[5 + i] = tp[(m + 16) * 5 + i]; }
    float* outp = knnbuf + ((size_t)cs * TPTS + (size_t)(R0 + m)) * KNN;
#pragma unroll
    for (int i = 0; i < KNN; ++i) {
      int best = i;
      for (int j = i + 1; j < 10; ++j) if (cand[j] < cand[best]) best = j;
      const float tv = cand[i]; cand[i] = cand[best]; cand[best] = tv;
      outp[i] = cand[i];
    }
  }
}

// ---------------- merge column splits, sqrt, deterministic partial sums ----------------
__global__ __launch_bounds__(256) void merge_kernel(const float* __restrict__ knnbuf,
                                                    float* __restrict__ psum,
                                                    float* __restrict__ psumsq) {
  __shared__ float rs1[256], rs2[256];
  const int row = blockIdx.x * 256 + threadIdx.x;
  float cand[CSPLIT * KNN];
#pragma unroll
  for (int c = 0; c < CSPLIT; ++c)
#pragma unroll
    for (int i = 0; i < KNN; ++i)
      cand[c * KNN + i] = knnbuf[((size_t)c * TPTS + row) * KNN + i];
  float s1 = 0.f, s2 = 0.f;
#pragma unroll
  for (int i = 0; i < KNN; ++i) {
    int best = i;
    for (int j = i + 1; j < CSPLIT * KNN; ++j) if (cand[j] < cand[best]) best = j;
    const float tv = cand[i]; cand[i] = cand[best]; cand[best] = tv;
    const float d = sqrtf(cand[i]);
    s1 += d; s2 += d * d;
  }
  rs1[threadIdx.x] = s1; rs2[threadIdx.x] = s2;
  __syncthreads();
  for (int off = 128; off > 0; off >>= 1) {
    if (threadIdx.x < off) { rs1[threadIdx.x] += rs1[threadIdx.x + off];
                             rs2[threadIdx.x] += rs2[threadIdx.x + off]; }
    __syncthreads();
  }
  if (threadIdx.x == 0) { psum[blockIdx.x] = rs1[0]; psumsq[blockIdx.x] = rs2[0]; }
}

__global__ void finalize_kernel(const float* __restrict__ psum,
                                const float* __restrict__ psumsq,
                                float* __restrict__ out) {
  if (threadIdx.x == 0 && blockIdx.x == 0) {
    double s1 = 0.0, s2 = 0.0;
    for (int i = 0; i < TPTS / 256; ++i) { s1 += (double)psum[i]; s2 += (double)psumsq[i]; }
    const double n = (double)TPTS * (double)KNN;
    const double mean = s1 / n;
    double var = (s2 - n * mean * mean) / (n - 1.0);
    if (var < 0.0) var = 0.0;
    out[0] = (float)(-sqrt(var));
  }
}

extern "C" void kernel_launch(void* const* d_in, const int* in_sizes, int n_in,
                              void* d_out, int out_size, void* d_ws, size_t ws_size,
                              hipStream_t stream) {
  (void)in_sizes; (void)n_in; (void)out_size; (void)ws_size;
  const float* latent = (const float*)d_in[0];   // [2][8192][512] f32; sample 0 = first plane
  char* ws = (char*)d_ws;

  const size_t offXhi = 0;
  const size_t offXlo = (size_t)TPTS * DIM * 2;                 // 8 MB
  const size_t offSq  = offXlo + (size_t)TPTS * DIM * 2;        // 16 MB
  const size_t offKnn = offSq + (size_t)TPTS * 4;               // +32 KB
  const size_t offP   = offKnn + (size_t)CSPLIT * TPTS * KNN * 4;

  unsigned short* xhi = (unsigned short*)(ws + offXhi);
  unsigned short* xlo = (unsigned short*)(ws + offXlo);
  float* sq     = (float*)(ws + offSq);
  float* knnbuf = (float*)(ws + offKnn);
  float* psum   = (float*)(ws + offP);
  float* psumsq = psum + (TPTS / 256);

  prep_kernel<<<TPTS / 8, 256, 0, stream>>>(latent, xhi, xlo, sq);
  knn_kernel<<<(TPTS / (16 * WPB)) * CSPLIT, 64, 0, stream>>>(xhi, xlo, sq, knnbuf);
  merge_kernel<<<TPTS / 256, 256, 0, stream>>>(knnbuf, psum, psumsq);
  finalize_kernel<<<1, 32, 0, stream>>>(psum, psumsq, (float*)d_out);
}
